// MultiHeadAttention_57157424775432
// MI455X (gfx1250) — compile-verified, matrix operand order checked by reading, not run on hardware
//
#include <hip/hip_runtime.h>
#include <hip/hip_bf16.h>
#include <cstdint>

typedef _Float16 v16h __attribute__((ext_vector_type(16)));
typedef _Float16 v8h  __attribute__((ext_vector_type(8)));
typedef float    v8f  __attribute__((ext_vector_type(8)));
typedef unsigned int u32x4 __attribute__((ext_vector_type(4)));
typedef int          i32x4 __attribute__((ext_vector_type(4)));
typedef int          i32x8 __attribute__((ext_vector_type(8)));
// exact pointee type of the async-LDS builtins per hipcc's diagnostic:
typedef int gv4i __attribute__((vector_size(16)));

#define S_LEN   2048
#define D_MODEL 1024
#define NH      16
#define DHEAD   64

#define AS1 __attribute__((address_space(1)))
#define AS3 __attribute__((address_space(3)))

// ---- CDNA5 feature detection (degrades to the plain-VMEM path if absent,
// ---- and on the host pass where the amdgcn builtins don't exist) ----------
#if defined(__has_builtin)
#if __has_builtin(__builtin_amdgcn_global_load_async_to_lds_b128)
#define MHA_HAVE_ASYNC 1
#endif
#if __has_builtin(__builtin_amdgcn_tensor_load_to_lds) && \
    __has_builtin(__builtin_amdgcn_s_wait_tensorcnt)
#define MHA_HAVE_TDM 1
#endif
#endif

#if defined(MHA_HAVE_ASYNC)
#if __has_builtin(__builtin_amdgcn_s_wait_asynccnt)
#define MHA_WAIT_ASYNC() __builtin_amdgcn_s_wait_asynccnt(0)
#else
#define MHA_WAIT_ASYNC() asm volatile("s_wait_asynccnt 0x0" ::: "memory")
#endif
#endif

// Load one 16x32 (f16) WMMA fragment row/column for this lane.
// gfx1250 16-bit layout: lanes 0-15 hold K={0..7,16..23}, lanes 16-31 hold
// K={8..15,24..31} -> two 16-byte chunks at offsets 8*sel and 16+8*sel halfs.
__device__ __forceinline__ v16h load_frag32(const _Float16* p, int lane) {
  const int sel = (lane >> 4) & 1;
  v8h c0 = *reinterpret_cast<const v8h*>(p + 8 * sel);
  v8h c1 = *reinterpret_cast<const v8h*>(p + 16 + 8 * sel);
  v16h r;
#pragma unroll
  for (int i = 0; i < 8; ++i) { r[i] = c0[i]; r[i + 8] = c1[i]; }
  return r;
}

// C = X @ W^T + bias ;  X: (4096 x 1024) TX, W: (1024 x 1024) f32 row-major.
// Tile: 128(M) x 64(N), K staged 32 at a time (converted to f16 in LDS).
template <typename TX>
__global__ __launch_bounds__(256) void gemm_xwT_kernel(
    const TX* __restrict__ X, const float* __restrict__ W,
    const float* __restrict__ bias, _Float16* __restrict__ outh,
    float* __restrict__ outf) {
  constexpr int TM = 128, TN = 64, TK = 32;
  __shared__ _Float16 lA[TM * TK];   // 8 KB
  __shared__ _Float16 lB[TN * TK];   // 4 KB
  const int tid  = threadIdx.x;
  const int lane = tid & 31;
  const int wid  = tid >> 5;
  const int m0   = blockIdx.x * TM;
  const int n0   = blockIdx.y * TN;

  v8f acc[4];
#pragma unroll
  for (int j = 0; j < 4; ++j)
#pragma unroll
    for (int r = 0; r < 8; ++r) acc[j][r] = 0.0f;

  for (int k0 = 0; k0 < D_MODEL; k0 += TK) {
    // ---- stage A tile ----
#if defined(MHA_HAVE_ASYNC)
    if constexpr (sizeof(TX) == 2) {
      // f16 input: pure copy -> async DMA straight into LDS (ASYNCcnt)
      for (int idx = tid; idx < TM * TK / 8; idx += 256) {
        const int row = idx >> 2, c = idx & 3;
        __builtin_amdgcn_global_load_async_to_lds_b128(
            (AS1 gv4i*)(&X[(size_t)(m0 + row) * D_MODEL + k0 + c * 8]),
            (AS3 gv4i*)(&lA[row * TK + c * 8]), 0, 0);
      }
    } else
#endif
    {
#pragma unroll 2
      for (int idx = tid; idx < TM * TK / 8; idx += 256) {   // 512 chunks of 8
        const int row = idx >> 2, c = idx & 3;
        const TX* src = &X[(size_t)(m0 + row) * D_MODEL + k0 + c * 8];
        _Float16* dst = &lA[row * TK + c * 8];
#pragma unroll
        for (int i = 0; i < 8; ++i) dst[i] = (_Float16)src[i];
      }
    }
    // ---- stage B tile (f32 -> f16 convert) ----
    for (int idx = tid; idx < TN * TK / 8; idx += 256) {     // 256 chunks of 8
      const int row = idx >> 2, c = idx & 3;
      const float* src = &W[(size_t)(n0 + row) * D_MODEL + k0 + c * 8];
      _Float16* dst = &lB[row * TK + c * 8];
#pragma unroll
      for (int i = 0; i < 8; ++i) dst[i] = (_Float16)src[i];
    }
#if defined(MHA_HAVE_ASYNC)
    if constexpr (sizeof(TX) == 2) MHA_WAIT_ASYNC();
#endif
    __syncthreads();

    v16h a = load_frag32(&lA[(wid * 16 + (lane & 15)) * TK], lane);
#pragma unroll
    for (int j = 0; j < 4; ++j) {
      v16h b = load_frag32(&lB[(j * 16 + (lane & 15)) * TK], lane);
      acc[j] = __builtin_amdgcn_wmma_f32_16x16x32_f16(
          false, a, false, b, (short)0, acc[j], false, false);
    }
    __syncthreads();
  }

  // C/D layout: vgpr r, lane L -> (M = r + 8*(L/16), N = L%16)
#pragma unroll
  for (int j = 0; j < 4; ++j) {
#pragma unroll
    for (int r = 0; r < 8; ++r) {
      const int m = m0 + wid * 16 + r + 8 * (lane >> 4);
      const int n = n0 + j * 16 + (lane & 15);
      const float v = acc[j][r] + bias[n];
      if (outh) outh[(size_t)m * D_MODEL + n] = (_Float16)v;
      else      outf[(size_t)m * D_MODEL + n] = v;
    }
  }
}

// Flash attention per (b, h, 128-query tile). The reference's double
// split_proj means Q row n reads Qp[b, (n%128)*16 + h, (n/128)*64 + d];
// the output transpose folds into the store: Aout[b, n, h*64+d].
__global__ __launch_bounds__(256) void attn_kernel(
    const _Float16* __restrict__ Qp, const _Float16* __restrict__ Kp,
    const _Float16* __restrict__ Vp, const int* __restrict__ mask,
    _Float16* __restrict__ Aout) {
  __shared__ _Float16 lK[64 * 64];        // K tile, key-major      (8 KB)
  __shared__ _Float16 lVt[64 * 64];       // V tile, d-major (T)    (8 KB)
  __shared__ _Float16 lP[8 * 16 * 64];    // per-wave P staging     (16 KB)
#if defined(MHA_HAVE_ASYNC)
  __shared__ _Float16 lV[64 * 64];        // V async staging        (8 KB)
#endif
  const int tid = threadIdx.x, lane = tid & 31, wid = tid >> 5;
  const int qt = blockIdx.x, h = blockIdx.y, b = blockIdx.z;
  const int q0 = qt * 128;

  // Q fragments (double-split permutation folded into the address)
  v16h qfrag[2];
  {
    const int n_local = wid * 16 + (lane & 15);                 // n % 128
    const size_t srow = (size_t)b * S_LEN + (size_t)n_local * NH + h;
    const int col = qt * DHEAD;                                 // (n/128)*64
#pragma unroll
    for (int kc = 0; kc < 2; ++kc)
      qfrag[kc] = load_frag32(&Qp[srow * D_MODEL + col + kc * 32], lane);
  }

  float mrow[8], lrow[8];
  v8f oacc[4];
#pragma unroll
  for (int r = 0; r < 8; ++r) { mrow[r] = -1e30f; lrow[r] = 0.0f; }
#pragma unroll
  for (int j = 0; j < 4; ++j)
#pragma unroll
    for (int r = 0; r < 8; ++r) oacc[j][r] = 0.0f;

  for (int kb = 0; kb < S_LEN; kb += 64) {
    const size_t gbase = ((size_t)b * S_LEN + kb) * D_MODEL + h * DHEAD;

    // ---------------- stage K tile ----------------
#if defined(MHA_HAVE_TDM)
    if (wid == 0) {
      // Tensor Data Mover: one D# moves the whole 64x64 f16 tile (TENSORcnt)
      const unsigned long long gK = (unsigned long long)(uintptr_t)(&Kp[gbase]);
      u32x4 g0;
      g0.x = 1u;                                        // count=1, user mode
      g0.y = (unsigned int)(uintptr_t)(&lK[0]);         // lds_addr
      g0.z = (unsigned int)(gK & 0xFFFFFFFFu);          // global_addr[31:0]
      g0.w = (unsigned int)(gK >> 32) | (2u << 30);     // addr[56:32] | type=2
      i32x8 g1;
      g1[0] = 0x00010000;          // wg_mask=0, data_size=1 (2 bytes)
      g1[1] = (1024 << 16);        // tensor_dim0 = 1024 (bits 79:48, low half)
      g1[2] = (2048 << 16);        // tensor_dim1 = 2048 (bits 111:80, low half)
      g1[3] = (64 << 16);          // tile_dim0 = 64     (bits 127:112)
      g1[4] = 64;                  // tile_dim1 = 64, tile_dim2 = 0
      g1[5] = 1024;                // tensor_dim0_stride = 1024 (low 32)
      g1[6] = 0;
      g1[7] = 0;
      i32x4 z4 = {0, 0, 0, 0};
#if __clang_major__ >= 23
      i32x8 z8 = {0, 0, 0, 0, 0, 0, 0, 0};
      __builtin_amdgcn_tensor_load_to_lds(g0, g1, z4, z4, z8, 0);
#else
      __builtin_amdgcn_tensor_load_to_lds(g0, g1, z4, z4, 0);
#endif
    }
#elif defined(MHA_HAVE_ASYNC)
    for (int idx = tid; idx < 512; idx += 256) {
      const int row = idx >> 3, c = idx & 7;
      __builtin_amdgcn_global_load_async_to_lds_b128(
          (AS1 gv4i*)(&Kp[gbase + (size_t)row * D_MODEL + c * 8]),
          (AS3 gv4i*)(&lK[row * 64 + c * 8]), 0, 0);
    }
#else
#pragma unroll 2
    for (int idx = tid; idx < 512; idx += 256) {
      const int row = idx >> 3, c = idx & 7;
      *reinterpret_cast<v8h*>(&lK[row * 64 + c * 8]) =
          *reinterpret_cast<const v8h*>(&Kp[gbase + (size_t)row * D_MODEL + c * 8]);
    }
#endif

    // ---------------- stage V tile ----------------
#if defined(MHA_HAVE_ASYNC)
    for (int idx = tid; idx < 512; idx += 256) {
      const int row = idx >> 3, c = idx & 7;
      __builtin_amdgcn_global_load_async_to_lds_b128(
          (AS1 gv4i*)(&Vp[gbase + (size_t)row * D_MODEL + c * 8]),
          (AS3 gv4i*)(&lV[row * 64 + c * 8]), 0, 0);
    }
    MHA_WAIT_ASYNC();
#else
    for (int idx = tid; idx < 512; idx += 256) {
      const int row = idx >> 3, c = idx & 7;
      v8h vv = *reinterpret_cast<const v8h*>(&Vp[gbase + (size_t)row * D_MODEL + c * 8]);
#pragma unroll
      for (int i = 0; i < 8; ++i) lVt[(c * 8 + i) * 64 + row] = vv[i];
    }
#endif
#if defined(MHA_HAVE_TDM)
    __builtin_amdgcn_s_wait_tensorcnt(0);
#endif
    __syncthreads();   // (A) staged tiles visible WG-wide

#if defined(MHA_HAVE_ASYNC)
    // LDS->LDS transpose of the async-staged V tile (DMA can't transpose);
    // overlaps with the QK^T WMMAs below, consumed only after barrier (B).
    for (int idx = tid; idx < 512; idx += 256) {
      const int row = idx >> 3, c = idx & 7;
      v8h vv = *reinterpret_cast<const v8h*>(&lV[row * 64 + c * 8]);
#pragma unroll
      for (int i = 0; i < 8; ++i) lVt[(c * 8 + i) * 64 + row] = vv[i];
    }
#endif

    // S = Q K^T * 1/8 + mask * -1e9
    float sc[4][8];
#pragma unroll
    for (int j = 0; j < 4; ++j) {
      v8f s;
#pragma unroll
      for (int r = 0; r < 8; ++r) s[r] = 0.0f;
#pragma unroll
      for (int kc = 0; kc < 2; ++kc) {
        v16h bfr = load_frag32(&lK[(j * 16 + (lane & 15)) * 64 + kc * 32], lane);
        s = __builtin_amdgcn_wmma_f32_16x16x32_f16(
            false, qfrag[kc], false, bfr, (short)0, s, false, false);
      }
      const int kcol = kb + j * 16 + (lane & 15);
#pragma unroll
      for (int r = 0; r < 8; ++r) {
        const int qrow = q0 + wid * 16 + r + 8 * (lane >> 4);
        const int mv = mask[(size_t)qrow * S_LEN + kcol];
        sc[j][r] = s[r] * 0.125f + (mv ? -1e9f : 0.0f);   // cndmask, not cvt
      }
    }

    // online softmax; row M lives in a 16-lane half -> width-16 reductions
#pragma unroll
    for (int r = 0; r < 8; ++r) {
      float mx = sc[0][r];
#pragma unroll
      for (int j = 1; j < 4; ++j) mx = fmaxf(mx, sc[j][r]);
#pragma unroll
      for (int off = 1; off < 16; off <<= 1) mx = fmaxf(mx, __shfl_xor(mx, off, 16));
      const float mnew  = fmaxf(mrow[r], mx);
      const float alpha = __expf(mrow[r] - mnew);
      mrow[r] = mnew;
      float rs = 0.0f;
#pragma unroll
      for (int j = 0; j < 4; ++j) {
        const float p = __expf(sc[j][r] - mnew);
        sc[j][r] = p;
        rs += p;
      }
#pragma unroll
      for (int off = 1; off < 16; off <<= 1) rs += __shfl_xor(rs, off, 16);
      lrow[r] = lrow[r] * alpha + rs;
#pragma unroll
      for (int j = 0; j < 4; ++j) oacc[j][r] *= alpha;
    }

    __syncthreads();   // (B) lVt transpose complete before PV reads

    // bounce P through LDS: C-layout -> A-fragment layout (same-wave only)
    _Float16* myP = &lP[wid * 16 * 64];
#pragma unroll
    for (int j = 0; j < 4; ++j)
#pragma unroll
      for (int r = 0; r < 8; ++r)
        myP[(r + 8 * (lane >> 4)) * 64 + j * 16 + (lane & 15)] = (_Float16)sc[j][r];

    v16h pf[2];
#pragma unroll
    for (int kc = 0; kc < 2; ++kc)
      pf[kc] = load_frag32(&myP[(lane & 15) * 64 + kc * 32], lane);

    // O += P @ V
#pragma unroll
    for (int j2 = 0; j2 < 4; ++j2) {
#pragma unroll
      for (int kc = 0; kc < 2; ++kc) {
        v16h bfr = load_frag32(&lVt[(j2 * 16 + (lane & 15)) * 64 + kc * 32], lane);
        oacc[j2] = __builtin_amdgcn_wmma_f32_16x16x32_f16(
            false, pf[kc], false, bfr, (short)0, oacc[j2], false, false);
      }
    }
    __syncthreads();   // (C) done with lK/lV/lVt before next stage
  }

  // finalize: divide by row sum, store in (B,S,D) layout (transpose folded)
#pragma unroll
  for (int j = 0; j < 4; ++j)
#pragma unroll
    for (int r = 0; r < 8; ++r) {
      const int m    = r + 8 * (lane >> 4);
      const int nrow = q0 + wid * 16 + m;
      const int d    = h * DHEAD + j * 16 + (lane & 15);
      const float o  = oacc[j][r] / lrow[r];
      Aout[((size_t)b * S_LEN + nrow) * D_MODEL + d] = (_Float16)o;
    }
}

extern "C" void kernel_launch(void* const* d_in, const int* in_sizes, int n_in,
                              void* d_out, int out_size, void* d_ws, size_t ws_size,
                              hipStream_t stream) {
  (void)in_sizes; (void)n_in; (void)out_size; (void)ws_size;
  const float* queries = (const float*)d_in[0];
  const float* keys    = (const float*)d_in[1];
  const float* values  = (const float*)d_in[2];
  const int*   mask    = (const int*)d_in[3];
  const float* Wq = (const float*)d_in[4];
  const float* bq = (const float*)d_in[5];
  const float* Wk = (const float*)d_in[6];
  const float* bk = (const float*)d_in[7];
  const float* Wv = (const float*)d_in[8];
  const float* bv = (const float*)d_in[9];
  const float* Wo = (const float*)d_in[10];
  const float* bo = (const float*)d_in[11];
  float* out = (float*)d_out;

  const size_t projBytes = (size_t)2 * S_LEN * D_MODEL * sizeof(_Float16); // 8 MB
  char* ws = (char*)d_ws;
  _Float16* Qp = (_Float16*)(ws + 0 * projBytes);
  _Float16* Kp = (_Float16*)(ws + 1 * projBytes);
  _Float16* Vp = (_Float16*)(ws + 2 * projBytes);
  _Float16* Ao = (_Float16*)(ws + 3 * projBytes);

  const dim3 blk(256);
  const dim3 ggrid(4096 / 128, D_MODEL / 64, 1);  // 32 x 16

  gemm_xwT_kernel<float><<<ggrid, blk, 0, stream>>>(queries, Wq, bq, Qp, nullptr);
  gemm_xwT_kernel<float><<<ggrid, blk, 0, stream>>>(keys,    Wk, bk, Kp, nullptr);
  gemm_xwT_kernel<float><<<ggrid, blk, 0, stream>>>(values,  Wv, bv, Vp, nullptr);

  const dim3 agrid(S_LEN / 128, NH, 2);           // 16 x 16 x 2
  attn_kernel<<<agrid, blk, 0, stream>>>(Qp, Kp, Vp, mask, Ao);

  gemm_xwT_kernel<_Float16><<<ggrid, blk, 0, stream>>>(Ao, Wo, bo, nullptr, out);
}
